// HyperbolicMean_32942399160957
// MI455X (gfx1250) — matrix operand order repaired and checked by reading, not compiled, not run on hardware
//
#include <hip/hip_runtime.h>

// Hyperbolic (Karcher) mean on the hyperboloid, N=131072 rows, d=512.
// Strategy: convert projected matrix P ([t,x], padded 513->544 cols) to f16 once
// (142.6 MB -> resident in MI455X 192 MB L2), then 100 iterations of:
//   pass_kernel  : WMMA row-dot GEMV (v_wmma_f32_16x16x32_f16) + per-row
//                  arccosh/sinh/cosh + weighted column sums (atomic f32)
//   update_kernel: 1-block exp-map update of the mean + accumulator reset
// Workspace (d_ws): P_f16 (N*544*2 B) | meanF32[544] | mprime_f16[544] | s1[544] | s2

typedef __attribute__((ext_vector_type(16))) _Float16 v16h;
typedef __attribute__((ext_vector_type(8)))  _Float16 v8h;
typedef __attribute__((ext_vector_type(8)))  float    v8f;

#define HM_D      512
#define HM_DP     513          // 1 + 512
#define HM_PAD    544          // 17 * 32, zero padded
#define HM_CH     17           // K chunks of 32
#define HM_RPB    128          // rows per block (8 waves * 16 rows)
#define HM_EPS    1e-5f
#define HM_R      0.01f
#define HM_ITERS  100

// ---------------------------------------------------------------- projection
// One block per row: t = sqrt(1 + |x|^2); write f16 row [t, x, 0-pad].
__global__ __launch_bounds__(256) void hm_project(const float* __restrict__ data,
                                                  _Float16* __restrict__ P) {
  __shared__ float red[256];
  const long row = blockIdx.x;
  const float* dr = data + row * (long)HM_D;
  _Float16* pr = P + row * (long)HM_PAD;
  const int tid = threadIdx.x;
  const float a0 = dr[2 * tid], a1 = dr[2 * tid + 1];
  red[tid] = a0 * a0 + a1 * a1;
  __syncthreads();
  for (int s = 128; s > 0; s >>= 1) {
    if (tid < s) red[tid] += red[tid + s];
    __syncthreads();
  }
  if (tid == 0) pr[0] = (_Float16)sqrtf(1.0f + red[0]);
  pr[1 + 2 * tid] = (_Float16)a0;
  pr[2 + 2 * tid] = (_Float16)a1;
  if (tid < HM_PAD - HM_DP) pr[HM_DP + tid] = (_Float16)0.0f;  // 31 pad cols
}

// ---------------------------------------------------------------- accumulator zero
__global__ void hm_zero(float* __restrict__ s1, float* __restrict__ s2) {
  const int t = blockIdx.x * blockDim.x + threadIdx.x;
  if (t < HM_PAD) s1[t] = 0.0f;
  if (t == 0) *s2 = 0.0f;
}

// ---------------------------------------------------------------- column sums of P
__global__ __launch_bounds__(256) void hm_colsum(const _Float16* __restrict__ P,
                                                 float* __restrict__ s1) {
  const long base = (long)blockIdx.x * HM_RPB;
  for (int j = threadIdx.x; j < HM_PAD; j += 256) {
    float a = 0.0f;
    const _Float16* cp = P + base * HM_PAD + j;
#pragma unroll 4
    for (int r = 0; r < HM_RPB; ++r) a += (float)cp[(long)r * HM_PAD];
    atomicAdd(&s1[j], a);
  }
}

// ---------------------------------------------------------------- init mean
// mean = colsum/N normalized by Minkowski norm; emit f32 mean + f16 m'; zero accum.
__global__ __launch_bounds__(1024) void hm_init(float* __restrict__ meanF,
                                                _Float16* __restrict__ mp,
                                                float* __restrict__ s1,
                                                float* __restrict__ s2, float Ninv) {
  __shared__ float red[1024];
  const int tid = threadIdx.x;
  float m = (tid < HM_DP) ? s1[tid] * Ninv : 0.0f;
  float t2 = m * m;
  if (tid == 0) t2 = -t2;  // Minkowski: sum(x^2) - 2*x0^2
  red[tid] = t2;
  __syncthreads();
  for (int s = 512; s > 0; s >>= 1) {
    if (tid < s) red[tid] += red[tid + s];
    __syncthreads();
  }
  const float norm = sqrtf(fabsf(red[0]));
  m /= norm;
  if (tid < HM_PAD) {
    meanF[tid] = (tid < HM_DP) ? m : 0.0f;
    mp[tid] = (_Float16)((tid == 0) ? -m : ((tid < HM_DP) ? m : 0.0f));
    s1[tid] = 0.0f;
  }
  if (tid == 0) *s2 = 0.0f;
}

// ---------------------------------------------------------------- main pass
// Phase 1: v_i = P_i . m' via v_wmma_f32_16x16x32_f16 (A = 16x32 row tile,
//          B = m' chunk replicated across all 16 columns -> D[:,0] = dots).
// Phase 2: per-row w = d/sinh(d), c = cosh(d); s1 += sum_r w_r * P_r
//          (coalesced weighted column sums), s2 += sum_r w_r * c_r.
__global__ __launch_bounds__(256) void hm_pass(const _Float16* __restrict__ P,
                                               const _Float16* __restrict__ mp,
                                               float* __restrict__ s1,
                                               float* __restrict__ s2) {
  __shared__ __align__(32) _Float16 mlds[HM_PAD];
  __shared__ float dots[HM_RPB];
  __shared__ float wbuf[HM_RPB];
  __shared__ float red[256];

  const int tid = threadIdx.x;
  for (int j = tid; j < HM_PAD; j += 256) mlds[j] = mp[j];
  __syncthreads();

  const int lane = tid & 31;
  const int wave = tid >> 5;
  const int g = lane >> 4;            // column-half group per A/B VGPR layout
  const int rowInWave = lane & 15;
  const long baseRow = (long)blockIdx.x * HM_RPB;
  const _Float16* rp = P + (baseRow + wave * 16 + rowInWave) * (long)HM_PAD;

  v8f acc = {};
#pragma unroll
  for (int c = 0; c < HM_CH; ++c) {
    const int kc = c * 32;
    // A-layout (16-bit, 16x32): elems 0-15 -> K = kc+16g+e (matches B)
    v16h a = *(const v16h*)(rp + kc + 16 * g);
    // B-layout (32x16): lane-group g holds K = kc+16g+e, identical for all columns
    v16h b = *(const v16h*)(&mlds[kc + 16 * g]);
    acc = __builtin_amdgcn_wmma_f32_16x16x32_f16(false, a, false, b,
                                                 (short)0, acc, false, false);
  }
  // D column 0: lane 0 holds rows 0-7 (VGPRs 0-7), lane 16 holds rows 8-15.
  if (lane == 0) {
#pragma unroll
    for (int r = 0; r < 8; ++r) dots[wave * 16 + r] = acc[r];
  } else if (lane == 16) {
#pragma unroll
    for (int r = 0; r < 8; ++r) dots[wave * 16 + 8 + r] = acc[r];
  }
  __syncthreads();

  float wc = 0.0f;
  if (tid < HM_RPB) {
    float x = -dots[tid];                       // -mdot(mean, p) = cosh(dist)
    x = fmaxf(x, 1.0f + HM_EPS);
    const float sq = sqrtf(x * x - 1.0f);
    const float d = logf(x) + log1pf(sq / x);   // arccosh(x)
    const float e = __expf(d);
    const float ei = 1.0f / e;
    const float ch = 0.5f * (e + ei);
    const float sh = 0.5f * (e - ei);
    const float w = (d > 1e-6f) ? d / sh : 1.0f;
    wbuf[tid] = w;
    wc = w * ch;
  }
  red[tid] = wc;
  __syncthreads();
  for (int s = 128; s > 0; s >>= 1) {
    if (tid < s) red[tid] += red[tid + s];
    __syncthreads();
  }
  if (tid == 0) atomicAdd(s2, red[0]);

  // Phase 2: weighted column sums (adjacent threads -> adjacent columns, coalesced)
  for (int j = tid; j < HM_PAD; j += 256) {
    float a = 0.0f;
    const _Float16* cp = P + baseRow * HM_PAD + j;
#pragma unroll 4
    for (int r = 0; r < HM_RPB; ++r) a += wbuf[r] * (float)cp[(long)r * HM_PAD];
    atomicAdd(&s1[j], a);
  }
}

// ---------------------------------------------------------------- exp-map update
__global__ __launch_bounds__(1024) void hm_update(float* __restrict__ meanF,
                                                  _Float16* __restrict__ mp,
                                                  float* __restrict__ s1,
                                                  float* __restrict__ s2, float Ninv) {
  __shared__ float red[1024];
  const int tid = threadIdx.x;
  const float s2v = *s2;
  const float m = (tid < HM_DP) ? meanF[tid] : 0.0f;
  const float s1v = (tid < HM_DP) ? s1[tid] : 0.0f;
  // g = -2/N (s1 - s2*mean);  y = -R*g
  const float y = (tid < HM_DP) ? (2.0f * HM_R * Ninv) * (s1v - s2v * m) : 0.0f;

  float t2 = y * y;
  if (tid == 0) t2 = -t2;
  red[tid] = t2;
  __syncthreads();
  for (int s = 512; s > 0; s >>= 1) {
    if (tid < s) red[tid] += red[tid + s];
    __syncthreads();
  }
  float n = sqrtf(fabsf(red[0]));
  n = fmaxf(n, HM_EPS);
  const float e = __expf(n), ei = 1.0f / e;
  const float ch = 0.5f * (e + ei), sh = 0.5f * (e - ei);
  float mn = ch * m + (sh / n) * y;             // exp_map

  float u2 = mn * mn;
  if (tid == 0) u2 = -u2;
  __syncthreads();
  red[tid] = u2;
  __syncthreads();
  for (int s = 512; s > 0; s >>= 1) {
    if (tid < s) red[tid] += red[tid + s];
    __syncthreads();
  }
  const float norm = sqrtf(fabsf(red[0]));
  mn /= norm;
  if (tid < HM_PAD) {
    meanF[tid] = (tid < HM_DP) ? mn : 0.0f;
    mp[tid] = (_Float16)((tid == 0) ? -mn : ((tid < HM_DP) ? mn : 0.0f));
    s1[tid] = 0.0f;                             // reset for next iteration
  }
  if (tid == 0) *s2 = 0.0f;
}

// ---------------------------------------------------------------- output
__global__ void hm_out(const float* __restrict__ meanF, float* __restrict__ out) {
  const int t = blockIdx.x * blockDim.x + threadIdx.x;
  if (t < HM_D) out[t] = meanF[1 + t];
}

extern "C" void kernel_launch(void* const* d_in, const int* in_sizes, int n_in,
                              void* d_out, int out_size, void* d_ws, size_t ws_size,
                              hipStream_t stream) {
  const float* data = (const float*)d_in[0];
  float* out = (float*)d_out;
  const long N = (long)in_sizes[0] / HM_D;      // 131072
  const float Ninv = 1.0f / (float)N;
  const int nblocks = (int)(N / HM_RPB);        // 1024

  char* ws = (char*)d_ws;
  _Float16* P = (_Float16*)ws;                  // N * 544 f16 (~142.6 MB)
  size_t off = (size_t)N * HM_PAD * sizeof(_Float16);
  off = (off + 255) & ~(size_t)255;
  float* meanF = (float*)(ws + off);  off += HM_PAD * sizeof(float);
  _Float16* mp = (_Float16*)(ws + off); off += HM_PAD * sizeof(_Float16);
  off = (off + 255) & ~(size_t)255;
  float* s1 = (float*)(ws + off);     off += HM_PAD * sizeof(float);
  off = (off + 255) & ~(size_t)255;
  float* s2 = (float*)(ws + off);

  hm_zero<<<(HM_PAD + 255) / 256, 256, 0, stream>>>(s1, s2);
  hm_project<<<(int)N, 256, 0, stream>>>(data, P);
  hm_colsum<<<nblocks, 256, 0, stream>>>(P, s1);
  hm_init<<<1, 1024, 0, stream>>>(meanF, mp, s1, s2, Ninv);
  for (int it = 0; it < HM_ITERS; ++it) {
    hm_pass<<<nblocks, 256, 0, stream>>>(P, mp, s1, s2);
    hm_update<<<1, 1024, 0, stream>>>(meanF, mp, s1, s2, Ninv);
  }
  hm_out<<<(HM_D + 255) / 256, 256, 0, stream>>>(meanF, out);
}
